// RoPEMultiHeadAttention_16269336117669
// MI455X (gfx1250) — compile-verified, matrix operand order checked by reading, not run on hardware
//
#include <hip/hip_runtime.h>

// ---------------------------------------------------------------------------
// RoPE multi-head attention for MI455X (gfx1250).
// bf16 WMMA everywhere, big-LDS fused attention, async global->LDS staging.
// ---------------------------------------------------------------------------

typedef __attribute__((ext_vector_type(16))) __bf16 v16bf;
typedef __attribute__((ext_vector_type(8)))  float  v8f;

#define D_MODEL 1024
#define N_HEADS 16
#define D_K     64
#define SEQ     2048
#define BATCH   2

// Generic pointer -> LDS byte offset (ptrtoint of the AS(3) pointer).
__device__ __forceinline__ unsigned lds_offset(const void* p) {
  return (unsigned)(unsigned long long)(__attribute__((address_space(3))) const void*)p;
}

// Load a 16(M/N) x 32(K) bf16 WMMA fragment from a row-major fp32 matrix.
// A-operand layout (ISA 7.12.2): lanes 0-15 row M=lane, K=0..7 & 16..23;
// lanes 16-31 same rows, K=8..15 & 24..31.  B-operand for C = X*W^T is loaded
// identically from rows of W (lane n holds row n of W == column n of W^T).
__device__ __forceinline__ v16bf load_frag_f32(const float* __restrict__ base, int ld) {
  const int lane  = threadIdx.x & 31;
  const int r     = lane & 15;
  const int halfk = lane >> 4;  // 0 or 1
  const float* p = base + (size_t)r * ld + halfk * 8;
  v16bf f;
#pragma unroll
  for (int i = 0; i < 8; ++i) f[i] = (__bf16)p[i];
#pragma unroll
  for (int i = 0; i < 8; ++i) f[8 + i] = (__bf16)p[16 + i];
  return f;
}

__device__ __forceinline__ v8f wmma_bf16(v16bf a, v16bf b, v8f c) {
  // (neg_a, A, neg_b, B, c_mod, C, reuse_a, reuse_b)
  return __builtin_amdgcn_wmma_f32_16x16x32_bf16(false, a, false, b, (short)0, c,
                                                 false, false);
}

// ---------------------------------------------------------------------------
// C[M,N] = X[M,K] * W[N,K]^T (+ optional bias[N]).
// Block = 256 threads = 8 waves (2x4), wave tile 32x32, block tile 64x128.
// ---------------------------------------------------------------------------
__global__ __launch_bounds__(256) void gemm_xwt_kernel(
    const float* __restrict__ X, const float* __restrict__ W,
    const float* __restrict__ bias, float* __restrict__ P,
    int N, int K) {
  const int wave = threadIdx.x >> 5;
  const int wm = wave >> 2;           // 0..1
  const int wn = wave & 3;            // 0..3
  const int m0 = blockIdx.x * 64 + wm * 32;
  const int n0 = blockIdx.y * 128 + wn * 32;

  v8f c00 = {}, c01 = {}, c10 = {}, c11 = {};
  for (int k = 0; k < K; k += 32) {
    if (k + 32 < K) {
      // W panels are hot in the 192 MB L2; prefetch next tile.
      __builtin_prefetch(W + (size_t)n0 * K + k + 32, 0, 0);
    }
    v16bf a0 = load_frag_f32(X + (size_t)m0 * K + k, K);
    v16bf a1 = load_frag_f32(X + (size_t)(m0 + 16) * K + k, K);
    v16bf b0 = load_frag_f32(W + (size_t)n0 * K + k, K);
    v16bf b1 = load_frag_f32(W + (size_t)(n0 + 16) * K + k, K);
    c00 = wmma_bf16(a0, b0, c00);
    c01 = wmma_bf16(a0, b1, c01);
    c10 = wmma_bf16(a1, b0, c10);
    c11 = wmma_bf16(a1, b1, c11);
  }

  const int lane   = threadIdx.x & 31;
  const int col    = lane & 15;
  const int rowoff = (lane >> 4) * 8;  // C layout: VGPR v -> M = v (+8 for hi lanes)
  const float bia0 = bias ? bias[n0 + col] : 0.0f;
  const float bia1 = bias ? bias[n0 + 16 + col] : 0.0f;
#pragma unroll
  for (int v = 0; v < 8; ++v) {
    const int m = rowoff + v;
    P[(size_t)(m0 + m) * N + n0 + col]           = c00[v] + bia0;
    P[(size_t)(m0 + m) * N + n0 + 16 + col]      = c01[v] + bia1;
    P[(size_t)(m0 + 16 + m) * N + n0 + col]      = c10[v] + bia0;
    P[(size_t)(m0 + 16 + m) * N + n0 + 16 + col] = c11[v] + bia1;
  }
}

// ---------------------------------------------------------------------------
// In-place RoPE on Q and K: for head h, pair (i, i+32), i in [0,32).
// One block per row of [B*S, D_MODEL].
// ---------------------------------------------------------------------------
__global__ __launch_bounds__(256) void rope_kernel(float* __restrict__ Q,
                                                   float* __restrict__ Kb) {
  const int row = blockIdx.x;         // 0..B*S-1
  const int s   = row & (SEQ - 1);
  for (int p = threadIdx.x; p < (D_MODEL / 2); p += 256) {
    const int h = p >> 5;
    const int i = p & 31;
    const float invf = __powf(10000.0f, -(float)i / 32.0f);
    float sn, cs;
    __sincosf((float)s * invf, &sn, &cs);
    const size_t i1 = (size_t)row * D_MODEL + h * D_K + i;
    const size_t i2 = i1 + 32;
    const float q1 = Q[i1], q2 = Q[i2];
    Q[i1] = q1 * cs - q2 * sn;
    Q[i2] = q2 * cs + q1 * sn;
    const float k1 = Kb[i1], k2 = Kb[i2];
    Kb[i1] = k1 * cs - k2 * sn;
    Kb[i2] = k2 * cs + k1 * sn;
  }
}

// ---------------------------------------------------------------------------
// Fused attention: block = (b*h, 32 query rows).  Score strip 32x2048 fp32
// lives entirely in LDS (256 KB — gfx1250 has 320 KB/WGP), so scores are
// computed, softmaxed, emitted to the attn output, and consumed by P@V
// without ever round-tripping through HBM.  V chunks are staged via the
// gfx1250 async global->LDS path (ASYNCcnt).
// ---------------------------------------------------------------------------
__global__ __launch_bounds__(256) void attn_kernel(
    const float* __restrict__ Q, const float* __restrict__ Kb,
    const float* __restrict__ V, float* __restrict__ attn,
    float* __restrict__ ctx) {
  extern __shared__ float smem[];
  float* sc     = smem;             // [32][2048] scores -> probs (256 KB)
  float* vstage = smem + 32 * 2048; // [64][64] V chunk staging   (16 KB)

  const int bh   = blockIdx.x;      // 0..31
  const int b    = bh >> 4;
  const int h    = bh & 15;
  const int q0   = blockIdx.y * 32;
  const size_t rowbase = (size_t)b * SEQ;
  const int wave = threadIdx.x >> 5;
  const int lane = threadIdx.x & 31;

  // ---- Phase 1: scores = (Q K^T) / sqrt(d_k) -> LDS ------------------------
  v16bf qf[2][2];
#pragma unroll
  for (int mt = 0; mt < 2; ++mt)
#pragma unroll
    for (int kf = 0; kf < 2; ++kf)
      qf[mt][kf] = load_frag_f32(
          Q + (rowbase + q0 + mt * 16) * D_MODEL + h * D_K + kf * 32, D_MODEL);

  for (int step = 0; step < 16; ++step) {
    const int c0 = wave * 256 + step * 16;  // key-column tile for this wave
    v16bf k0f = load_frag_f32(Kb + (rowbase + c0) * D_MODEL + h * D_K, D_MODEL);
    v16bf k1f = load_frag_f32(Kb + (rowbase + c0) * D_MODEL + h * D_K + 32, D_MODEL);
#pragma unroll
    for (int mt = 0; mt < 2; ++mt) {
      v8f acc = {};
      acc = wmma_bf16(qf[mt][0], k0f, acc);
      acc = wmma_bf16(qf[mt][1], k1f, acc);
      const int col    = lane & 15;
      const int rowoff = (lane >> 4) * 8;
#pragma unroll
      for (int v = 0; v < 8; ++v)
        sc[(mt * 16 + rowoff + v) * 2048 + c0 + col] = acc[v] * 0.125f;
    }
  }
  __syncthreads();

  // ---- Phase 2: row softmax in LDS, emit attn to HBM -----------------------
  for (int rr = 0; rr < 4; ++rr) {
    const int r = wave * 4 + rr;
    float* srow = sc + (size_t)r * 2048;
    float m = -3.4e38f;
    for (int c = lane; c < 2048; c += 32) m = fmaxf(m, srow[c]);
#pragma unroll
    for (int off = 16; off > 0; off >>= 1) m = fmaxf(m, __shfl_xor(m, off, 32));
    float sum = 0.0f;
    for (int c = lane; c < 2048; c += 32) {
      const float e = __expf(srow[c] - m);
      srow[c] = e;
      sum += e;
    }
#pragma unroll
    for (int off = 16; off > 0; off >>= 1) sum += __shfl_xor(sum, off, 32);
    const float inv = 1.0f / sum;
    float* grow = attn + ((size_t)bh * SEQ + q0 + r) * SEQ;
    for (int c = lane; c < 2048; c += 32) {
      const float pv = srow[c] * inv;
      srow[c] = pv;   // kept in LDS for the P@V phase
      grow[c] = pv;   // the mandatory 537 MB output write
    }
  }
  __syncthreads();

  // ---- Phase 3: ctx = P @ V ------------------------------------------------
  // Each wave owns one 16x16 output tile (2 m-tiles x 4 n-tiles of d_k=64).
  const int mt    = wave >> 2;
  const int nt    = wave & 3;
  const int ncol  = nt * 16 + (lane & 15);
  const int halfk = lane >> 4;
  const unsigned vbase = lds_offset(vstage);
  v8f acc = {};
  for (int kc = 0; kc < SEQ; kc += 64) {
    __syncthreads();
    // Async-copy V[kc..kc+63][head dims] (16 KB) straight into LDS,
    // bypassing VGPRs: 1024 x b128 transfers tracked by ASYNCcnt.
    for (int i = threadIdx.x; i < 1024; i += 256) {
      const int r  = i >> 4;          // 0..63 (V row within chunk)
      const int c4 = (i & 15) * 4;    // fp32 column group of 4
      const float* gp = V + (rowbase + kc + r) * D_MODEL + h * D_K + c4;
      const unsigned loff = vbase + (unsigned)i * 16u;
      asm volatile("global_load_async_to_lds_b128 %0, %1, off"
                   :: "v"(loff), "v"(gp) : "memory");
    }
    asm volatile("s_wait_asynccnt 0x0" ::: "memory");
    __syncthreads();
#pragma unroll
    for (int kk = 0; kk < 2; ++kk) {
      // A fragment straight from the LDS probability strip
      v16bf a = load_frag_f32(sc + (size_t)(mt * 16) * 2048 + kc + kk * 32, 2048);
      // B fragment: lane n gathers column n of the staged V chunk (on-chip)
      const float* vs = vstage + (kk * 32) * 64;
      v16bf bfr;
#pragma unroll
      for (int i = 0; i < 8; ++i) bfr[i] = (__bf16)vs[(halfk * 8 + i) * 64 + ncol];
#pragma unroll
      for (int i = 0; i < 8; ++i) bfr[8 + i] = (__bf16)vs[(halfk * 8 + 16 + i) * 64 + ncol];
      acc = wmma_bf16(a, bfr, acc);
    }
  }
  const int col    = lane & 15;
  const int rowoff = (lane >> 4) * 8;
#pragma unroll
  for (int v = 0; v < 8; ++v)
    ctx[(rowbase + q0 + mt * 16 + rowoff + v) * D_MODEL + h * D_K + nt * 16 + col] = acc[v];
}

// ---------------------------------------------------------------------------
// Host launcher.
// Inputs: query, key, value, Wq, Wk, Wv, Wo, bo (all fp32).
// d_out = out[2,2048,1024] ++ attn[2,16,2048,2048] (fp32).
// ---------------------------------------------------------------------------
extern "C" void kernel_launch(void* const* d_in, const int* in_sizes, int n_in,
                              void* d_out, int out_size, void* d_ws, size_t ws_size,
                              hipStream_t stream) {
  (void)in_sizes; (void)n_in; (void)out_size; (void)ws_size;
  const float* query = (const float*)d_in[0];
  const float* key   = (const float*)d_in[1];
  const float* value = (const float*)d_in[2];
  const float* Wq    = (const float*)d_in[3];
  const float* Wk    = (const float*)d_in[4];
  const float* Wv    = (const float*)d_in[5];
  const float* Wo    = (const float*)d_in[6];
  const float* bo    = (const float*)d_in[7];

  float* out  = (float*)d_out;                                 // [2,2048,1024]
  float* attn = out + (size_t)BATCH * SEQ * D_MODEL;           // [2,16,2048,2048]

  const size_t mat = (size_t)BATCH * SEQ * D_MODEL;            // 4M floats
  float* Qb  = (float*)d_ws;
  float* Kbf = Qb + mat;
  float* Vb  = Kbf + mat;
  float* ctx = Vb + mat;

  const dim3 blk(256);
  const dim3 gemm_grid((BATCH * SEQ) / 64, D_MODEL / 128);     // (64, 8)

  gemm_xwt_kernel<<<gemm_grid, blk, 0, stream>>>(query, Wq, nullptr, Qb, D_MODEL, D_MODEL);
  gemm_xwt_kernel<<<gemm_grid, blk, 0, stream>>>(key,   Wk, nullptr, Kbf, D_MODEL, D_MODEL);
  gemm_xwt_kernel<<<gemm_grid, blk, 0, stream>>>(value, Wv, nullptr, Vb, D_MODEL, D_MODEL);

  rope_kernel<<<dim3(BATCH * SEQ), blk, 0, stream>>>(Qb, Kbf);

  const size_t shmem = (size_t)(32 * 2048 + 64 * 64) * sizeof(float); // 272 KB < 320 KB/WGP
  attn_kernel<<<dim3(BATCH * N_HEADS, SEQ / 32), blk, shmem, stream>>>(Qb, Kbf, Vb, attn, ctx);

  gemm_xwt_kernel<<<gemm_grid, blk, 0, stream>>>(ctx, Wo, bo, out, D_MODEL, D_MODEL);
}
